// VectorQuantizer_1297080123930
// MI455X (gfx1250) — matrix operand order repaired and checked by reading, not compiled, not run on hardware
//
#include <hip/hip_runtime.h>

// Problem constants (reference: K=4096 codes, D=256 dims, N=32768 latents)
#define DD   256
#define KK   4096
#define BETA 0.25f

// Tiling
#define MT    64          // latent rows per block
#define CT    32          // codebook rows per LDS tile
#define XSTR  264         // padded LDS row stride in bf16 elements (breaks bank stride)

typedef __attribute__((ext_vector_type(16))) __bf16          v16bf;
typedef __attribute__((ext_vector_type(8)))  float           v8f;
typedef __attribute__((ext_vector_type(8)))  unsigned short  ush8;

// Matches the builtin's expected pointee: int __attribute__((vector_size(16)))
typedef int b128_t __attribute__((vector_size(16)));

union Frag {
    struct { ush8 lo, hi; } h;
    v16bf v;
};

__device__ __forceinline__ unsigned short f32_to_bf16_rne(float f) {
    unsigned u = __float_as_uint(f);
    u += 0x7FFFu + ((u >> 16) & 1u);     // round-to-nearest-even
    return (unsigned short)(u >> 16);
}

// 16-byte global->LDS copy. Prefer the gfx1250 async-to-LDS DMA path
// (GLOBAL_LOAD_ASYNC_TO_LDS_B128, tracked by ASYNCcnt); fall back to a
// VGPR round-trip if the toolchain lacks the builtin.
__device__ __forceinline__ void cp16_g2l(unsigned short* ldst,
                                         const unsigned short* gsrc) {
#if defined(__gfx1250__) && __has_builtin(__builtin_amdgcn_global_load_async_to_lds_b128)
    __builtin_amdgcn_global_load_async_to_lds_b128(
        (__attribute__((address_space(1))) b128_t*)gsrc,
        (__attribute__((address_space(3))) b128_t*)ldst,
        0, 0);
#else
    *(ush8*)ldst = *(const ush8*)gsrc;
#endif
}

__device__ __forceinline__ void cp_async_wait_all() {
#if defined(__gfx1250__) && __has_builtin(__builtin_amdgcn_s_wait_asynccnt)
    __builtin_amdgcn_s_wait_asynccnt(0);
#elif defined(__gfx1250__) && __has_builtin(__builtin_amdgcn_global_load_async_to_lds_b128)
    asm volatile("s_wait_asynccnt 0x0" ::: "memory");
#endif
}

// ---------------------------------------------------------------------------
// Kernel 1: one wave per code — compute ||e||^2 (fp32) AND convert the
// codebook row to bf16 (done ONCE, instead of per-block).
// ---------------------------------------------------------------------------
__global__ __launch_bounds__(256) void vq_prep_kernel(const float* __restrict__ cb,
                                                      float* __restrict__ enorm,
                                                      unsigned short* __restrict__ ebf) {
    const int lane = threadIdx.x & 31;
    const int code = blockIdx.x * 8 + (threadIdx.x >> 5);
    const float* row = cb + (size_t)code * DD;
    unsigned short* orow = ebf + (size_t)code * DD;
    float s = 0.f;
    #pragma unroll
    for (int d = lane; d < DD; d += 32) {
        const float v = row[d];
        s += v * v;
        orow[d] = f32_to_bf16_rne(v);
    }
    #pragma unroll
    for (int m = 16; m; m >>= 1) s += __shfl_xor(s, m, 32);
    if (lane == 0) enorm[code] = s;
}

// ---------------------------------------------------------------------------
// Kernel 2: fused bf16-WMMA codebook search + exact fp32 epilogue.
// ---------------------------------------------------------------------------
__global__ __launch_bounds__(256) void vq_main_kernel(const float* __restrict__ X,
                                                      const float* __restrict__ E,
                                                      const unsigned short* __restrict__ Ebf,
                                                      const float* __restrict__ enorm,
                                                      float* __restrict__ out_q,
                                                      float* __restrict__ out_loss) {
    __shared__ unsigned short sX[MT * XSTR];   // 33792 B: latent tile, bf16
    __shared__ unsigned short sE[CT * XSTR];   // 16896 B: codebook tile, bf16
    __shared__ float sNorm[CT];
    __shared__ float sMin[2][MT];
    __shared__ int   sIdx[2][MT];

    const int tid  = threadIdx.x;
    const int lane = tid & 31;
    const int wave = tid >> 5;
    const int half = lane >> 4;        // 0: lanes 0-15, 1: lanes 16-31
    const int l16  = lane & 15;
    const int rowBase = blockIdx.x * MT;

    // ---- Phase 0: stage X tile (64 x 256) to LDS as bf16 (once per block) ----
    for (int i = tid; i < MT * DD; i += 256) {
        const int r = i >> 8, d = i & 255;
        sX[r * XSTR + d] = f32_to_bf16_rne(X[(size_t)(rowBase + r) * DD + d]);
    }

    // Wave roles: 4 row-slabs x 2 column-halves of the CT=32 code tile.
    const int slab = wave & 3;         // rows slab*16 .. slab*16+15
    const int chal = wave >> 2;        // columns chal*16 .. chal*16+15

    float bmin[8];
    int   bidx[8];
    #pragma unroll
    for (int v = 0; v < 8; ++v) { bmin[v] = __builtin_inff(); bidx[v] = 0; }

    const int aRow = slab * 16 + l16;
    const int bCol = chal * 16 + l16;

    for (int ct = 0; ct < KK / CT; ++ct) {
        __syncthreads();               // previous tile's WMMA reads done before overwrite
        // ---- DMA codebook tile (32 x 256 bf16 = 16 KB) into LDS ----
        // 1024 chunks of 16 B; 32 chunks per code row (512 B / row, exact).
        const unsigned short* gE = Ebf + (size_t)ct * CT * DD;
        #pragma unroll
        for (int j = tid; j < CT * DD / 8; j += 256) {
            const int c = j >> 5;          // code within tile
            const int w = j & 31;          // 16B chunk within row
            cp16_g2l(&sE[c * XSTR + w * 8], gE + j * 8);
        }
        if (tid < CT) sNorm[tid] = enorm[ct * CT + tid];
        // prefetch next tile toward L2/L0 while we compute (global_prefetch_b8)
        if (ct + 1 < KK / CT)
            __builtin_prefetch(Ebf + (size_t)(ct + 1) * CT * DD + tid * 32, 0, 0);
        cp_async_wait_all();
        __syncthreads();

        // ---- 16x16 dot-product tile: acc = X_slab . E_colgroup^T over D=256 ----
        v8f acc = {0.f, 0.f, 0.f, 0.f, 0.f, 0.f, 0.f, 0.f};
        #pragma unroll
        for (int kk = 0; kk < DD; kk += 32) {
            // A fragment (16x32 bf16): lane half 0 holds K {kk..kk+7, kk+16..kk+23},
            // half 1 holds K {kk+8..kk+15, kk+24..kk+31}  (ISA 7.12.2 layout)
            const int aoff = aRow * XSTR + kk + (half ? 8 : 0);
            Frag A; A.h.lo = *(const ush8*)&sX[aoff];
                    A.h.hi = *(const ush8*)&sX[aoff + 16];
            // B fragment (32x16 bf16): lane half 0 holds K {kk..kk+7, kk+16..kk+23} for col l16,
            // half 1 holds K {kk+8..kk+15, kk+24..kk+31} — same order as A
            const int boff = bCol * XSTR + kk + (half ? 8 : 0);
            Frag B; B.h.lo = *(const ush8*)&sE[boff];
                    B.h.hi = *(const ush8*)&sE[boff + 16];
            acc = __builtin_amdgcn_wmma_f32_16x16x32_bf16(
                      false, A.v, false, B.v, (short)0, acc, false, false);
        }

        // ---- epilogue: score = ||e||^2 - 2*dot ; running argmin ----
        const float en     = sNorm[chal * 16 + l16];
        const int   colIdx = ct * CT + chal * 16 + l16;
        #pragma unroll
        for (int v = 0; v < 8; ++v) {
            const float sc = __builtin_fmaf(-2.0f, acc[v], en);
            if (sc < bmin[v]) { bmin[v] = sc; bidx[v] = colIdx; }
        }
    }

    // ---- reduce argmin across the 16 lanes of each half (each holds one column) ----
    #pragma unroll
    for (int v = 0; v < 8; ++v) {
        float s = bmin[v]; int i = bidx[v];
        #pragma unroll
        for (int m = 1; m < 16; m <<= 1) {
            const float so = __shfl_xor(s, m, 32);
            const int   io = __shfl_xor(i, m, 32);
            if (so < s || (so == s && io < i)) { s = so; i = io; }
        }
        if (l16 == 0) {                 // lane 0 -> rows v, lane 16 -> rows v+8
            const int r = slab * 16 + v + 8 * half;
            sMin[chal][r] = s;
            sIdx[chal][r] = i;
        }
    }
    __syncthreads();

    // ---- merge the two column-halves per row ----
    if (tid < MT) {
        const float s0 = sMin[0][tid], s1 = sMin[1][tid];
        const int   i0 = sIdx[0][tid], i1 = sIdx[1][tid];
        sIdx[0][tid] = (s1 < s0 || (s1 == s0 && i1 < i0)) ? i1 : i0;
    }
    __syncthreads();

    // ---- Phase 2: exact fp32 outputs. One wave per 8 rows. ----
    for (int r = wave * 8; r < wave * 8 + 8; ++r) {
        const int grow = rowBase + r;
        const int code = sIdx[0][r];
        const float* xr = X + (size_t)grow * DD;
        const float* er = E + (size_t)code * DD;
        float part = 0.f;
        #pragma unroll
        for (int d = lane; d < DD; d += 32) {
            const float x = xr[d];
            const float q = er[d];
            const float diff = q - x;
            out_q[(size_t)grow * DD + d] = x + diff;   // straight-through: x + (q - x)
            part += diff * diff;
        }
        #pragma unroll
        for (int m = 16; m; m >>= 1) part += __shfl_xor(part, m, 32);
        if (lane == 0) out_loss[grow] = (1.0f + BETA) * part;  // emb + beta*commit
    }
}

// ---------------------------------------------------------------------------
extern "C" void kernel_launch(void* const* d_in, const int* in_sizes, int n_in,
                              void* d_out, int out_size, void* d_ws, size_t ws_size,
                              hipStream_t stream) {
    const float* latents  = (const float*)d_in[0];   // [N, 256] fp32
    const float* codebook = (const float*)d_in[1];   // [4096, 256] fp32
    const int N = in_sizes[0] / DD;

    float* out_q    = (float*)d_out;                 // [N, 256]
    float* out_loss = out_q + (size_t)N * DD;        // [N]

    // workspace: [4096] fp32 norms, then [4096*256] bf16 codebook (2 MB)
    float*          enorm = (float*)d_ws;
    unsigned short* ebf   = (unsigned short*)(enorm + KK);

    vq_prep_kernel<<<KK / 8, 256, 0, stream>>>(codebook, enorm, ebf);
    vq_main_kernel<<<N / MT, 256, 0, stream>>>(latents, codebook, ebf, enorm,
                                               out_q, out_loss);
}
